// PlaneAttention3D_32186484916607
// MI455X (gfx1250) — matrix operand order repaired and checked by reading, not compiled, not run on hardware
//
#include <hip/hip_runtime.h>
#include <hip/hip_bf16.h>

// ---------------------------------------------------------------------------
// PlaneAttention3D for MI455X (gfx1250, wave32, WMMA).
// Identity: full-token softmax attention is permutation-equivariant, so the
// xy/xz/yz branches are equal => ONE attention pass: Wp*attn(x)+bp+x.
// All GEMMs: v_wmma_f32_16x16x32_bf16 (bf16 operands, f32 accumulate).
// GEMM waves compute 32x64 strips (8 WMMA per 6 fragment loads per K-step).
// Flash attention: register-resident online softmax (shfl_xor reductions on
// the D-fragment layout), double-buffered K-frags, early-issued V-frags,
// wave-local LDS only (no block barriers).
// ---------------------------------------------------------------------------

typedef __attribute__((ext_vector_type(16))) __bf16 v16bf;
typedef __attribute__((ext_vector_type(8)))  float  v8f;

#define BB 2
#define CC 256
#define NN 4096
#define NH 4
#define DH 64

__device__ __forceinline__ __bf16 f2bf(float f) {
  union { float f; unsigned u; } a; a.f = f;
  unsigned r = a.u + 0x7FFFu + ((a.u >> 16) & 1u);   // round-to-nearest-even
  unsigned short h = (unsigned short)(r >> 16);
  __bf16 b; __builtin_memcpy(&b, &h, 2); return b;
}

__device__ __forceinline__ v8f wmma_bf16(v16bf a, v16bf b, v8f c) {
  return __builtin_amdgcn_wmma_f32_16x16x32_bf16(
      false, a, false, b, (short)0, c, false, false);
}

// A-fragment (16x32 bf16): lane holds row M=lane%16; two 16B chunks:
//   chunk0 -> K = kbase + 16*(lane/16) + [0..7]
//   chunk1 -> K = kbase + 16*(lane/16) + 8 + [0..7]  (contiguous k, same as B)
union Frag16 { v16bf v; uint4 q[2]; };

// ---------------------------------------------------------------------------
// Prep: fp32 -> bf16 conversions / transpose
// ---------------------------------------------------------------------------
__global__ void k_convert(const float* __restrict__ src,
                          __bf16* __restrict__ dst, int n) {
  int i = blockIdx.x * blockDim.x + threadIdx.x;
  if (i < n) dst[i] = f2bf(src[i]);
}

// xt[b][n][c] = bf16(x[b][c][n])  (token-major for B-fragment loads)
__global__ void k_transpose_cast(const float* __restrict__ x,
                                 __bf16* __restrict__ xt) {
  int i = blockIdx.x * blockDim.x + threadIdx.x;  // i = (b*NN + n)*CC + c
  int c = i & (CC - 1);
  int n = (i >> 8) & (NN - 1);
  int b = i >> 20;
  xt[i] = f2bf(x[((size_t)b * CC + c) * NN + n]);
}

// ---------------------------------------------------------------------------
// GEMM1: qkv = Wqkv @ x.  M=768 K=256 N=4096 per batch.
// Wave computes a 32(o) x 64(n) strip: 8 WMMA per K-step of 32.
// Q (scaled) / K token-major [b,h,n,64]; V channel-major [b,h,64,n].
// ---------------------------------------------------------------------------
__global__ void __launch_bounds__(128)
k_gemm_qkv(const __bf16* __restrict__ wqkv, const __bf16* __restrict__ xt,
           __bf16* __restrict__ qb, __bf16* __restrict__ kb,
           __bf16* __restrict__ vb) {
  const int lane = threadIdx.x & 31, wave = threadIdx.x >> 5;
  const int hf = lane >> 4, l16 = lane & 15;
  const int b = blockIdx.z;
  const int obase = blockIdx.y * 128 + wave * 32;
  const int nbase = blockIdx.x * 64;

  const __bf16* aptr[2] = { wqkv + (size_t)(obase + l16) * CC,
                            wqkv + (size_t)(obase + 16 + l16) * CC };
  const __bf16* xrow = xt + (size_t)b * NN * CC;

  auto loadA = [&](int kt, Frag16 (&a)[2]) {
    #pragma unroll
    for (int ma = 0; ma < 2; ++ma) {
      a[ma].q[0] = *(const uint4*)(aptr[ma] + kt * 32 + 16 * hf);
      a[ma].q[1] = *(const uint4*)(aptr[ma] + kt * 32 + 16 * hf + 8);
    }
  };
  auto loadB = [&](int kt, v16bf (&bfr)[4]) {
    #pragma unroll
    for (int nb = 0; nb < 4; ++nb) {
      int n = nbase + nb * 16 + l16;
      bfr[nb] = *(const v16bf*)(xrow + (size_t)n * CC + kt * 32 + 16 * hf);
    }
  };

  v8f acc[2][4] = {};
  Frag16 a[2][2]; v16bf bfr[2][4];
  loadA(0, a[0]); loadB(0, bfr[0]);
  #pragma unroll
  for (int kt = 0; kt < 8; ++kt) {          // K = 256, step 32; ping-pong regs
    const int cur = kt & 1, nxt = cur ^ 1;
    if (kt < 7) { loadA(kt + 1, a[nxt]); loadB(kt + 1, bfr[nxt]); }
    #pragma unroll
    for (int ma = 0; ma < 2; ++ma)
      #pragma unroll
      for (int nb = 0; nb < 4; ++nb)
        acc[ma][nb] = wmma_bf16(a[cur][ma].v, bfr[cur][nb], acc[ma][nb]);
  }

  #pragma unroll
  for (int ma = 0; ma < 2; ++ma) {
    const int orow0 = obase + ma * 16;
    const int which = orow0 >> 8;           // 0=q 1=k 2=v (strip stays inside)
    const int h = (orow0 >> 6) & 3;
    const int bh = b * NH + h;
    #pragma unroll
    for (int nb = 0; nb < 4; ++nb) {
      int n = nbase + nb * 16 + l16;
      #pragma unroll
      for (int r = 0; r < 8; ++r) {
        int o = orow0 + 8 * hf + r;
        int dd = o & 63;
        float v = acc[ma][nb][r];
        if (which == 0)      qb[((size_t)bh * NN + n) * DH + dd] = f2bf(v * 0.125f);
        else if (which == 1) kb[((size_t)bh * NN + n) * DH + dd] = f2bf(v);
        else                 vb[((size_t)bh * DH + dd) * NN + n] = f2bf(v);
      }
    }
  }
}

// ---------------------------------------------------------------------------
// Flash attention: grid (64 qtiles, 4 heads, 2 batch), 128 thr = 4 waves.
// Wave owns 16 query rows. Per 64-key tile: 8 WMMA (S) + 8 WMMA (O), online
// softmax fully in registers; K-frags double-buffered, V-frags early-issued.
// No __syncthreads: all LDS traffic is wave-local (per-wave DS is in-order).
// ---------------------------------------------------------------------------
__global__ void __launch_bounds__(128)
k_flash(const __bf16* __restrict__ qb, const __bf16* __restrict__ kb,
        const __bf16* __restrict__ vb, __bf16* __restrict__ ob) {
  __shared__ __bf16 sP[4][16][64];   // per-wave P strip (D-layout -> A-layout)

  const int lane = threadIdx.x & 31, wave = threadIdx.x >> 5;
  const int hf = lane >> 4, l16 = lane & 15;
  const int h = blockIdx.y, b = blockIdx.z;
  const int bh = b * NH + h;
  const int qrow = blockIdx.x * 64 + wave * 16;

  // Q fragments (loop invariant): rows = query tokens, K-dim = head dim
  Frag16 aq[2];
  {
    const __bf16* qp = qb + ((size_t)bh * NN + qrow + l16) * DH;
    #pragma unroll
    for (int ks = 0; ks < 2; ++ks) {
      aq[ks].q[0] = *(const uint4*)(qp + ks * 32 + 16 * hf);
      aq[ks].q[1] = *(const uint4*)(qp + ks * 32 + 16 * hf + 8);
    }
  }

  const __bf16* kbase = kb + (size_t)bh * NN * DH;
  const __bf16* vbase = vb + (size_t)bh * DH * NN;

  auto loadK = [&](int kt, v16bf (&kf)[8]) {     // B-frags for S = Q^T K
    #pragma unroll
    for (int nb = 0; nb < 4; ++nb) {
      const __bf16* kp = kbase + (size_t)(kt * 64 + nb * 16 + l16) * DH;
      #pragma unroll
      for (int ks = 0; ks < 2; ++ks)
        kf[nb * 2 + ks] = *(const v16bf*)(kp + ks * 32 + 16 * hf);
    }
  };

  v8f o[4] = {};
  float mrow[8], lrow[8];
  #pragma unroll
  for (int r = 0; r < 8; ++r) { mrow[r] = -1e30f; lrow[r] = 0.0f; }

  v16bf kA[8], kB[8];
  loadK(0, kA);

  auto tile = [&](int kt, v16bf (&kc)[8], v16bf (&kn)[8]) {
    // early-issue V B-frags (latency hidden behind softmax)
    v16bf vf[8];
    #pragma unroll
    for (int ks = 0; ks < 2; ++ks)
      #pragma unroll
      for (int nb = 0; nb < 4; ++nb)
        vf[ks * 4 + nb] = *(const v16bf*)(vbase + (size_t)(nb * 16 + l16) * NN
                                          + kt * 64 + ks * 32 + 16 * hf);
    // prefetch key rows two tiles ahead (global_prefetch_b8)
    if (kt + 2 < 64)
      __builtin_prefetch(kbase + (size_t)((kt + 2) * 64 + lane * 2) * DH, 0, 1);

    // S = Q^T K
    v8f s[4] = {};
    #pragma unroll
    for (int nb = 0; nb < 4; ++nb)
      #pragma unroll
      for (int ks = 0; ks < 2; ++ks)
        s[nb] = wmma_bf16(aq[ks].v, kc[nb * 2 + ks], s[nb]);

    // double-buffer: next tile's K frags in flight during softmax
    if (kt + 1 < 64) loadK(kt + 1, kn);

    // ---- online softmax in registers on the D-fragment layout ------------
    // row M = 8*hf + r lives in the 16 lanes of this half-group; xor masks
    // 1/2/4/8 reduce across those lanes; nb handled in-lane.
    #pragma unroll
    for (int r = 0; r < 8; ++r) {
      float t = fmaxf(fmaxf(s[0][r], s[1][r]), fmaxf(s[2][r], s[3][r]));
      t = fmaxf(t, __shfl_xor(t, 1, 32));
      t = fmaxf(t, __shfl_xor(t, 2, 32));
      t = fmaxf(t, __shfl_xor(t, 4, 32));
      t = fmaxf(t, __shfl_xor(t, 8, 32));
      float mnew  = fmaxf(mrow[r], t);
      float alpha = __expf(mrow[r] - mnew);
      float ls = 0.0f;
      #pragma unroll
      for (int nb = 0; nb < 4; ++nb) {
        float p = __expf(s[nb][r] - mnew);
        s[nb][r] = p;                       // s now holds P (f32)
        ls += p;
      }
      ls += __shfl_xor(ls, 1, 32);
      ls += __shfl_xor(ls, 2, 32);
      ls += __shfl_xor(ls, 4, 32);
      ls += __shfl_xor(ls, 8, 32);
      lrow[r] = lrow[r] * alpha + ls;
      mrow[r] = mnew;
      #pragma unroll
      for (int nb = 0; nb < 4; ++nb) o[nb][r] *= alpha;   // in-lane rescale
    }

    // D-layout -> A-layout transpose of P via wave-local LDS (no barrier:
    // same wave's DS ops are in-order)
    #pragma unroll
    for (int nb = 0; nb < 4; ++nb)
      #pragma unroll
      for (int r = 0; r < 8; ++r)
        sP[wave][8 * hf + r][nb * 16 + l16] = f2bf(s[nb][r]);

    // O += P * V^T
    #pragma unroll
    for (int ks = 0; ks < 2; ++ks) {
      Frag16 ap;
      const __bf16* pp = &sP[wave][l16][0];
      ap.q[0] = *(const uint4*)(pp + ks * 32 + 16 * hf);
      ap.q[1] = *(const uint4*)(pp + ks * 32 + 16 * hf + 8);
      #pragma unroll
      for (int nb = 0; nb < 4; ++nb)
        o[nb] = wmma_bf16(ap.v, vf[ks * 4 + nb], o[nb]);
    }
  };

  for (int kt = 0; kt < 64; kt += 2) {      // ping-pong K fragment buffers
    tile(kt,     kA, kB);
    tile(kt + 1, kB, kA);
  }

  // finalize: in-lane 1/l normalization, store token-major bf16
  #pragma unroll
  for (int r = 0; r < 8; ++r) {
    float inv = 1.0f / lrow[r];
    int tok = qrow + 8 * hf + r;
    #pragma unroll
    for (int nb = 0; nb < 4; ++nb) {
      int c = h * DH + nb * 16 + l16;
      ob[((size_t)b * NN + tok) * CC + c] = f2bf(o[nb][r] * inv);
    }
  }
}

// ---------------------------------------------------------------------------
// GEMM3: out[b,c,n] = Wp[c,:] . attn_out[:,n] + bp[c] + x[b,c,n]  (fp32 out)
// Wave computes a 32(c) x 64(n) strip.
// ---------------------------------------------------------------------------
__global__ void __launch_bounds__(128)
k_gemm_proj(const __bf16* __restrict__ wp, const __bf16* __restrict__ ob,
            const float* __restrict__ bp, const float* __restrict__ x,
            float* __restrict__ out) {
  const int lane = threadIdx.x & 31, wave = threadIdx.x >> 5;
  const int hf = lane >> 4, l16 = lane & 15;
  const int b = blockIdx.z;
  const int cbase = blockIdx.y * 128 + wave * 32;
  const int nbase = blockIdx.x * 64;

  const __bf16* aptr[2] = { wp + (size_t)(cbase + l16) * CC,
                            wp + (size_t)(cbase + 16 + l16) * CC };
  const __bf16* orow = ob + (size_t)b * NN * CC;

  auto loadA = [&](int kt, Frag16 (&a)[2]) {
    #pragma unroll
    for (int ma = 0; ma < 2; ++ma) {
      a[ma].q[0] = *(const uint4*)(aptr[ma] + kt * 32 + 16 * hf);
      a[ma].q[1] = *(const uint4*)(aptr[ma] + kt * 32 + 16 * hf + 8);
    }
  };
  auto loadB = [&](int kt, v16bf (&bfr)[4]) {
    #pragma unroll
    for (int nb = 0; nb < 4; ++nb) {
      int n = nbase + nb * 16 + l16;
      bfr[nb] = *(const v16bf*)(orow + (size_t)n * CC + kt * 32 + 16 * hf);
    }
  };

  v8f acc[2][4] = {};
  Frag16 a[2][2]; v16bf bfr[2][4];
  loadA(0, a[0]); loadB(0, bfr[0]);
  #pragma unroll
  for (int kt = 0; kt < 8; ++kt) {
    const int cur = kt & 1, nxt = cur ^ 1;
    if (kt < 7) { loadA(kt + 1, a[nxt]); loadB(kt + 1, bfr[nxt]); }
    #pragma unroll
    for (int ma = 0; ma < 2; ++ma)
      #pragma unroll
      for (int nb = 0; nb < 4; ++nb)
        acc[ma][nb] = wmma_bf16(a[cur][ma].v, bfr[cur][nb], acc[ma][nb]);
  }

  #pragma unroll
  for (int ma = 0; ma < 2; ++ma) {
    #pragma unroll
    for (int nb = 0; nb < 4; ++nb) {
      int n = nbase + nb * 16 + l16;
      #pragma unroll
      for (int r = 0; r < 8; ++r) {
        int c = cbase + ma * 16 + 8 * hf + r;
        size_t idx = ((size_t)b * CC + c) * NN + n;
        out[idx] = acc[ma][nb][r] + bp[c] + x[idx];
      }
    }
  }
}

// ---------------------------------------------------------------------------
extern "C" void kernel_launch(void* const* d_in, const int* in_sizes, int n_in,
                              void* d_out, int out_size, void* d_ws, size_t ws_size,
                              hipStream_t stream) {
  const float* x    = (const float*)d_in[0];   // [2,256,16,16,16]
  const float* Wqkv = (const float*)d_in[1];   // [768,256]
  const float* Wp   = (const float*)d_in[2];   // [256,256]
  const float* bp   = (const float*)d_in[3];   // [256]
  float* out = (float*)d_out;

  char* w = (char*)d_ws;
  auto carve = [&](size_t bytes) {
    char* p = w; w += (bytes + 255) & ~(size_t)255; return p;
  };
  __bf16* xt    = (__bf16*)carve((size_t)BB * NN * CC * 2);        // 4 MB
  __bf16* wqkvb = (__bf16*)carve((size_t)3 * CC * CC * 2);
  __bf16* wpb   = (__bf16*)carve((size_t)CC * CC * 2);
  __bf16* qb    = (__bf16*)carve((size_t)BB * NH * NN * DH * 2);   // 4 MB
  __bf16* kb    = (__bf16*)carve((size_t)BB * NH * NN * DH * 2);
  __bf16* vb    = (__bf16*)carve((size_t)BB * NH * NN * DH * 2);
  __bf16* ob    = (__bf16*)carve((size_t)BB * NN * CC * 2);

  k_transpose_cast<<<(BB * NN * CC) / 256, 256, 0, stream>>>(x, xt);
  k_convert<<<(3 * CC * CC) / 256, 256, 0, stream>>>(Wqkv, wqkvb, 3 * CC * CC);
  k_convert<<<(CC * CC) / 256, 256, 0, stream>>>(Wp, wpb, CC * CC);

  k_gemm_qkv<<<dim3(NN / 64, 768 / 128, BB), 128, 0, stream>>>(wqkvb, xt, qb, kb, vb);
  k_flash<<<dim3(NN / 64, NH, BB), 128, 0, stream>>>(qb, kb, vb, ob);
  k_gemm_proj<<<dim3(NN / 64, CC / 128, BB), 128, 0, stream>>>(wpb, ob, bp, x, out);
}
